// PolicyGNN_31095563223240
// MI455X (gfx1250) — matrix operand order repaired and checked by reading, not compiled, not run on hardware
//
#include <hip/hip_runtime.h>

typedef __attribute__((ext_vector_type(16))) _Float16 v16h;
typedef __attribute__((ext_vector_type(8)))  float    v8f;

// ---------------------------------------------------------------------------
// Degree / normalization kernels
// ---------------------------------------------------------------------------
__global__ __launch_bounds__(256)
void init_deg_kernel(float* __restrict__ deg, int n) {
  int i = blockIdx.x * blockDim.x + threadIdx.x;
  if (i < n) deg[i] = 1.0f;                       // self-loop contributes 1
}

__global__ __launch_bounds__(256)
void deg_scatter_kernel(const long long* __restrict__ ei, float* __restrict__ deg,
                        long long E) {
  long long e = (long long)blockIdx.x * blockDim.x + threadIdx.x;
  if (e < E) {
    int d = (int)ei[E + e];                       // dst row
    atomicAdd(deg + d, 1.0f);
  }
}

__global__ __launch_bounds__(256)
void rsqrt_kernel(float* __restrict__ deg, int n) {
  int i = blockIdx.x * blockDim.x + threadIdx.x;
  if (i < n) deg[i] = rsqrtf(deg[i]);             // deg >= 1 always
}

// ---------------------------------------------------------------------------
// Weight packing: f32 [K x nb] row-major -> f16 WMMA B-fragments.
// Fragment layout (contiguous k order used by this code, 16-bit B 32x16): lane group g = lane>>4 holds
// 16 consecutive K values at k0 + 16*g for column (lane&15).
// Packed index: ((tn*KSTEPS + ks)*32 + lane)*16 + j   -> one 32B load per lane.
// ---------------------------------------------------------------------------
__global__ __launch_bounds__(256)
void pack_weights_kernel(const float* __restrict__ B, _Float16* __restrict__ P,
                         int nb, int K) {
  const int ksteps = K >> 5;
  const int total  = (nb >> 4) * ksteps * 32;
  int t = blockIdx.x * blockDim.x + threadIdx.x;
  if (t >= total) return;
  const int lane = t & 31;
  const int rest = t >> 5;
  const int ks = rest % ksteps;
  const int tn = rest / ksteps;
  const int col   = tn * 16 + (lane & 15);
  const int kbase = ks * 32 + (lane >> 4) * 16;
  _Float16* out = P + (size_t)t * 16;
#pragma unroll
  for (int j = 0; j < 16; ++j)
    out[j] = (_Float16)B[(size_t)(kbase + j) * nb + col];
}

// ---------------------------------------------------------------------------
// WMMA GEMM with N register blocking:
// one wave computes a 16 x (16*NT) strip of C, loading each A fragment once
// per k-step and issuing NT independent back-to-back WMMAs.
// A: f32 row-major (float4 loads, f32->f16 cvt, optional fused ReLU)
// B: pre-packed f16 fragments (one 32B v16h load per tile per k-step per lane)
// M is an exact multiple of 16 (EXEC all-1s for WMMA).
// MODE 0 (GCN): H = C ; G = bias + C * dinv[row]^2  (self-loop + bias init)
// MODE 1 (FC):  H = tanh(C + bias)
// MODE 2 (out): H = C + bias
// ---------------------------------------------------------------------------
__device__ __forceinline__ float4 relu4(float4 v) {
  v.x = fmaxf(v.x, 0.0f); v.y = fmaxf(v.y, 0.0f);
  v.z = fmaxf(v.z, 0.0f); v.w = fmaxf(v.w, 0.0f);
  return v;
}

template<int K, int NT, int RELU, int MODE>
__global__ __launch_bounds__(256)
void wmma_gemm_kernel(const float* __restrict__ A, int lda,
                      const v16h* __restrict__ Bp, int nb,
                      int mtiles,
                      const float* __restrict__ bias,
                      const float* __restrict__ dinv,
                      float* __restrict__ H, float* __restrict__ G)
{
  constexpr int KSTEPS = K / 32;
  const int groups = (nb >> 4) / NT;              // N-tile groups per row strip
  const int wid = blockIdx.x * (blockDim.x >> 5) + (threadIdx.x >> 5);
  const int tm = wid / groups;
  const int tg = wid - tm * groups;
  if (tm >= mtiles) return;                       // wave-uniform early out

  const int lane = threadIdx.x & 31;
  const int half = lane >> 4;                     // lane group
  const int idx  = lane & 15;

  // A fragment base: row tm*16+idx, K offset half*16 within each 32-wide step
  const float* Arow = A + (size_t)(tm * 16 + idx) * lda + half * 16;
  // B fragment bases for the NT tiles of this group
  const v16h* Bfrag[NT];
#pragma unroll
  for (int i = 0; i < NT; ++i)
    Bfrag[i] = Bp + (size_t)(tg * NT + i) * KSTEPS * 32 + lane;

  v8f acc[NT];
#pragma unroll
  for (int i = 0; i < NT; ++i) acc[i] = (v8f){};

#pragma unroll
  for (int ks = 0; ks < KSTEPS; ++ks) {
    const float4* Ap = (const float4*)(Arow + ks * 32);
    float4 x0 = Ap[0];                            // K: kb+0 .. kb+3
    float4 x1 = Ap[1];                            // K: kb+4 .. kb+7
    float4 x2 = Ap[2];                            // K: kb+8 .. kb+11
    float4 x3 = Ap[3];                            // K: kb+12 .. kb+15
    if (RELU) { x0 = relu4(x0); x1 = relu4(x1); x2 = relu4(x2); x3 = relu4(x3); }

    v16h a;
    a[0]  = (_Float16)x0.x; a[1]  = (_Float16)x0.y;
    a[2]  = (_Float16)x0.z; a[3]  = (_Float16)x0.w;
    a[4]  = (_Float16)x1.x; a[5]  = (_Float16)x1.y;
    a[6]  = (_Float16)x1.z; a[7]  = (_Float16)x1.w;
    a[8]  = (_Float16)x2.x; a[9]  = (_Float16)x2.y;
    a[10] = (_Float16)x2.z; a[11] = (_Float16)x2.w;
    a[12] = (_Float16)x3.x; a[13] = (_Float16)x3.y;
    a[14] = (_Float16)x3.z; a[15] = (_Float16)x3.w;

#pragma unroll
    for (int i = 0; i < NT; ++i) {
      v16h b = Bfrag[i][ks * 32];                 // 32B coalesced fragment load
      acc[i] = __builtin_amdgcn_wmma_f32_16x16x32_f16(
          /*neg_a=*/false, a, /*neg_b=*/false, b,
          /*c_mod=*/(short)0, acc[i], /*reuse_a=*/false, /*reuse_b=*/false);
    }
  }

  // C layout: VGPR v -> M = v + 8*half ; N = lane & 15
#pragma unroll
  for (int i = 0; i < NT; ++i) {
    const int col = (tg * NT + i) * 16 + idx;
#pragma unroll
    for (int v = 0; v < 8; ++v) {
      const int r = tm * 16 + v + half * 8;
      const size_t o = (size_t)r * nb + col;
      const float val = acc[i][v];
      if (MODE == 0) {
        H[o] = val;                               // h = x @ W (gathered by edges)
        const float di = dinv[r];
        G[o] = bias[col] + val * di * di;         // bias + self-loop term
      } else if (MODE == 1) {
        H[o] = tanhf(val + bias[col]);
      } else {
        H[o] = val + bias[col];
      }
    }
  }
}

// ---------------------------------------------------------------------------
// Edge aggregation: G[dst] += norm(e) * H[src], norm = dinv[src]*dinv[dst].
// One thread per edge, 64 channels via float4 gathers (L2-resident H/G).
// ---------------------------------------------------------------------------
__global__ __launch_bounds__(256)
void edge_scatter_kernel(const long long* __restrict__ ei,
                         const float* __restrict__ h,
                         const float* __restrict__ dinv,
                         float* __restrict__ g, long long E)
{
  long long e = (long long)blockIdx.x * blockDim.x + threadIdx.x;
  if (e >= E) return;
  const int s = (int)ei[e];
  const int d = (int)ei[E + e];
  const float nrm = dinv[s] * dinv[d];
  const float4* hs = (const float4*)(h + (size_t)s * 64);
  float* gd = g + (size_t)d * 64;
#pragma unroll
  for (int i = 0; i < 16; ++i) {
    float4 v = hs[i];
    atomicAdd(gd + 4 * i + 0, v.x * nrm);
    atomicAdd(gd + 4 * i + 1, v.y * nrm);
    atomicAdd(gd + 4 * i + 2, v.z * nrm);
    atomicAdd(gd + 4 * i + 3, v.w * nrm);
  }
}

// ---------------------------------------------------------------------------
extern "C" void kernel_launch(void* const* d_in, const int* in_sizes, int n_in,
                              void* d_out, int out_size, void* d_ws, size_t ws_size,
                              hipStream_t stream)
{
  const float*     obs  = (const float*)d_in[0];
  const long long* ei   = (const long long*)d_in[1];
  const float*     W0   = (const float*)d_in[2];
  const float*     b0   = (const float*)d_in[3];
  const float*     Ws   = (const float*)d_in[4];
  const float*     bs   = (const float*)d_in[5];
  const float*     fc1W = (const float*)d_in[6];
  const float*     fc1b = (const float*)d_in[7];
  const float*     fc2W = (const float*)d_in[8];
  const float*     fc2b = (const float*)d_in[9];
  const float*     lgW  = (const float*)d_in[10];
  const float*     lgb  = (const float*)d_in[11];

  const int OBS = 128, HID = 64, FC = 256, OUT = 16, NGCN = 10;
  const int N = in_sizes[0] / OBS;                // 100000
  const long long E = in_sizes[1] / 2;            // 1600000
  const int mtiles = (N + 15) / 16;               // 6250 (exact)

  // Workspace carving (all buffers fully overwritten every call)
  char* p = (char*)d_ws;
  auto carve = [&](size_t bytes) -> char* {
    char* r = p; p += (bytes + 255) & ~(size_t)255; return r;
  };
  float* dinv = (float*)carve((size_t)N * sizeof(float));
  float* H    = (float*)carve((size_t)N * HID * sizeof(float));
  float* G0   = (float*)carve((size_t)N * HID * sizeof(float));
  float* G1   = (float*)carve((size_t)N * HID * sizeof(float));
  float* F1   = (float*)carve((size_t)N * FC * sizeof(float));
  float* F2   = (float*)carve((size_t)N * FC * sizeof(float));
  _Float16* PW0  = (_Float16*)carve((size_t)OBS * HID * 2);
  _Float16* PWs  = (_Float16*)carve((size_t)(NGCN - 1) * HID * HID * 2);
  _Float16* Pfc1 = (_Float16*)carve((size_t)HID * FC * 2);
  _Float16* Pfc2 = (_Float16*)carve((size_t)FC * FC * 2);
  _Float16* Plg  = (_Float16*)carve((size_t)FC * OUT * 2);

  const int nblk = (N + 255) / 256;
  const int eblk = (int)((E + 255) / 256);

  // Symmetric normalization: dinv = rsqrt(1 + in-degree)
  init_deg_kernel<<<nblk, 256, 0, stream>>>(dinv, N);
  deg_scatter_kernel<<<eblk, 256, 0, stream>>>(ei, dinv, E);
  rsqrt_kernel<<<nblk, 256, 0, stream>>>(dinv, N);

  // Pack all weights into f16 fragment order (tiny, L2-hot thereafter)
  auto pack = [&](const float* B, _Float16* P, int nb, int K) {
    const int total = (nb / 16) * (K / 32) * 32;
    pack_weights_kernel<<<(total + 255) / 256, 256, 0, stream>>>(B, P, nb, K);
  };
  pack(W0, PW0, HID, OBS);
  for (int l = 1; l < NGCN; ++l)
    pack(Ws + (size_t)(l - 1) * HID * HID, PWs + (size_t)(l - 1) * HID * HID,
         HID, HID);
  pack(fc1W, Pfc1, FC, HID);
  pack(fc2W, Pfc2, FC, FC);
  pack(lgW, Plg, OUT, FC);

  auto blocks = [&](int nb, int nt) {
    return (mtiles * ((nb / 16) / nt) + 7) / 8;   // 8 waves per 256-thread block
  };

  // GCN layer 0: obs (N x 128) @ W0 (128 x 64)
  wmma_gemm_kernel<128, 4, 0, 0><<<blocks(HID, 4), 256, 0, stream>>>(
      obs, OBS, (const v16h*)PW0, HID, mtiles, b0, dinv, H, G0);
  edge_scatter_kernel<<<eblk, 256, 0, stream>>>(ei, H, dinv, G0, E);

  // GCN layers 1..9 (ReLU of previous layer fused into A-fragment load)
  float* X = G0; float* Gn = G1;
  for (int l = 1; l < NGCN; ++l) {
    const v16h*  Wl = (const v16h*)(PWs + (size_t)(l - 1) * HID * HID);
    const float* bl = bs + (size_t)(l - 1) * HID;
    wmma_gemm_kernel<64, 4, 1, 0><<<blocks(HID, 4), 256, 0, stream>>>(
        X, HID, Wl, HID, mtiles, bl, dinv, H, Gn);
    edge_scatter_kernel<<<eblk, 256, 0, stream>>>(ei, H, dinv, Gn, E);
    float* t = X; X = Gn; Gn = t;
  }

  // FC head: tanh(relu(X) @ fc1W + b) -> tanh(.. @ fc2W + b) -> logits
  wmma_gemm_kernel<64, 4, 1, 1><<<blocks(FC, 4), 256, 0, stream>>>(
      X, HID, (const v16h*)Pfc1, FC, mtiles, fc1b, dinv, F1, nullptr);
  wmma_gemm_kernel<256, 4, 0, 1><<<blocks(FC, 4), 256, 0, stream>>>(
      F1, FC, (const v16h*)Pfc2, FC, mtiles, fc2b, dinv, F2, nullptr);
  wmma_gemm_kernel<256, 1, 0, 2><<<blocks(OUT, 1), 256, 0, stream>>>(
      F2, FC, (const v16h*)Plg, OUT, mtiles, lgb, dinv, (float*)d_out, nullptr);
}
